// EncoderLayer_49572512530890
// MI455X (gfx1250) — matrix operand order repaired and checked by reading, not compiled, not run on hardware
//
#include <hip/hip_runtime.h>

// ---------------------------------------------------------------------------
// Transformer encoder layer for MI455X (gfx1250, wave32, WMMA + TDM).
// B=4, S=2048, D=1024, H=16, DH=64, FFN=4096. fp32 in/out, bf16 WMMA compute.
// ---------------------------------------------------------------------------

#define BB 4
#define SS 2048
#define DD 1024
#define HH 16
#define DHH 64
#define FFN 4096
#define MM (BB * SS)        // 8192 rows
#define EPS 1e-5f

typedef __attribute__((ext_vector_type(16))) __bf16   v16bf;
typedef __attribute__((ext_vector_type(8)))  float    v8f;
typedef __attribute__((ext_vector_type(8)))  unsigned short ushort8;
typedef __attribute__((ext_vector_type(4)))  unsigned int   uint4v;
typedef __attribute__((ext_vector_type(8)))  int            int8v;
typedef __attribute__((ext_vector_type(4)))  int            int4v;

#if defined(__has_builtin)
#if __has_builtin(__builtin_amdgcn_tensor_load_to_lds) && \
    __has_builtin(__builtin_amdgcn_s_wait_tensorcnt)
#define USE_TDM 1
#endif
#endif
#ifndef USE_TDM
#define USE_TDM 0
#endif

union FragU {
    v16bf   bf;
    ushort8 us[2];
};

__device__ __forceinline__ unsigned short f32_to_bf16(float f) {
    unsigned int u = __builtin_bit_cast(unsigned int, f);
    u += 0x7FFFu + ((u >> 16) & 1u);   // round-to-nearest-even
    return (unsigned short)(u >> 16);
}

__device__ __forceinline__ v8f wmma_bf16(const v16bf a, const v16bf b, const v8f c) {
    return __builtin_amdgcn_wmma_f32_16x16x32_bf16(
        /*neg_a=*/false, a, /*neg_b=*/false, b,
        /*c_mod=*/(short)0, c, /*reuse_a=*/false, /*reuse_b=*/false);
}

__device__ __forceinline__ unsigned lds_off(const void* p) {
    // Generic pointer to __shared__: low 32 bits carry the LDS byte offset.
    return (unsigned)(unsigned long long)(size_t)p;
}

#if USE_TDM
// Issue a TDM copy of a 128-row x 64-element bf16 tile (row stride = K elems)
// from global memory into LDS. D# layout per CDNA5 ISA ch. 8 (2-D tensor,
// groups 2/3 zero; trailing group zero = terminating null descriptor).
// Wave-level instruction: call from one wave only.
__device__ __forceinline__ void tdm_load_tile_128x64(
        const unsigned short* gtile,   // global address of tile start
        unsigned ldsByteOff,           // LDS destination byte offset
        unsigned kElems) {             // tensor dim0 length == row stride
    const unsigned long long ga = (unsigned long long)(size_t)gtile;
    uint4v g0;
    g0[0] = 1u;                                             // count=1 (valid), user mode
    g0[1] = ldsByteOff;                                     // lds_addr
    g0[2] = (unsigned)(ga & 0xffffffffu);                   // global_addr[31:0]
    g0[3] = (unsigned)((ga >> 32) & 0x1ffffffu)             // global_addr[56:32]
            | 0x80000000u;                                  // type=2 ("image")
    int8v g1;
    g1[0] = (int)(1u << 16);                                // data_size=1 -> 2 bytes
    g1[1] = (int)((kElems & 0xffffu) << 16);                // tensor_dim0[15:0]
    g1[2] = (int)(((kElems >> 16) & 0xffffu)                // tensor_dim0[31:16]
            | (128u << 16));                                // tensor_dim1[15:0] = 128 rows
    g1[3] = (int)(64u << 16);                               // tensor_dim1[31:16]=0, tile_dim0=64
    g1[4] = (int)128u;                                      // tile_dim1=128, tile_dim2=0
    g1[5] = (int)kElems;                                    // tensor_dim0_stride[31:0]
    g1[6] = 0;                                              // stride hi / dim1_stride lo
    g1[7] = 0;
    const int4v z4 = {0, 0, 0, 0};
    const int8v z8 = {0, 0, 0, 0, 0, 0, 0, 0};
    __builtin_amdgcn_tensor_load_to_lds(g0, g1, z4, z4, z8, 0);
}
#endif

// ---------------------------------------------------------------------------
// fp32 -> bf16 cast
// ---------------------------------------------------------------------------
__global__ __launch_bounds__(256) void cast_bf16_kernel(const float* __restrict__ in,
                                                        unsigned short* __restrict__ out,
                                                        int n) {
    int i = blockIdx.x * 256 + threadIdx.x;
    if (i < n) out[i] = f32_to_bf16(in[i]);
}

// ---------------------------------------------------------------------------
// GEMM: Y[M,N] = X[M,K] @ W[N,K]^T + bias[N]  (bf16 inputs, f32 accum)
// Block tile 128x128, 8 waves (wave tile 64x32), K-step 64 through
// double-buffered LDS fed by the Tensor Data Mover.
// ---------------------------------------------------------------------------
__global__ __launch_bounds__(256) void gemm_bf16_kernel(
        const unsigned short* __restrict__ A,   // [M,K] bf16
        const unsigned short* __restrict__ W,   // [N,K] bf16
        const float* __restrict__ bias,         // [N]
        float* __restrict__ outF,               // [M,N] or null
        unsigned short* __restrict__ outB,      // [M,N] bf16 or null
        int M, int N, int K, int relu) {
    __shared__ __align__(16) unsigned short As[2][128 * 64];
    __shared__ __align__(16) unsigned short Bs[2][128 * 64];

    const int tid  = threadIdx.x;
    const int wave = tid >> 5;
    const int lane = tid & 31;
    const int lr   = lane & 15;
    const int hi   = lane >> 4;

    const int N0 = blockIdx.x * 128;
    const int M0 = blockIdx.y * 128;

    const int wm = (wave & 1) * 64;   // wave M offset in block
    const int wn = (wave >> 1) * 32;  // wave N offset in block

    v8f acc[4][2];
#pragma unroll
    for (int mi = 0; mi < 4; ++mi)
#pragma unroll
        for (int ni = 0; ni < 2; ++ni)
#pragma unroll
            for (int r = 0; r < 8; ++r) acc[mi][ni][r] = 0.0f;

    const unsigned short* Abase = A + (size_t)M0 * K;
    const unsigned short* Wbase = W + (size_t)N0 * K;
    const int kIters = K >> 6;   // 64 elements per iteration

#if USE_TDM
    if (wave == 0) {
        tdm_load_tile_128x64(Abase, lds_off(&As[0][0]), (unsigned)K);
        tdm_load_tile_128x64(Wbase, lds_off(&Bs[0][0]), (unsigned)K);
    }
#endif

    for (int it = 0; it < kIters; ++it) {
        const int cur = it & 1;
        const int nxt = cur ^ 1;
        const int kt  = it << 6;

#if USE_TDM
        __syncthreads();   // everyone finished reading buffer 'nxt' (iter it-1)
        if (wave == 0) {
            if (it + 1 < kIters) {
                tdm_load_tile_128x64(Abase + kt + 64, lds_off(&As[nxt][0]), (unsigned)K);
                tdm_load_tile_128x64(Wbase + kt + 64, lds_off(&Bs[nxt][0]), (unsigned)K);
                __builtin_amdgcn_s_wait_tensorcnt(2);  // current pair complete
            } else {
                __builtin_amdgcn_s_wait_tensorcnt(0);
            }
        }
        __syncthreads();   // publish TDM-written tiles to all waves
#else
        // Synchronous fallback staging (no TDM builtin available).
        __syncthreads();
        {
            const int row = tid >> 1;
            const int c0  = (tid & 1) * 32;
            const unsigned short* ga = Abase + (size_t)row * K + kt + c0;
            const unsigned short* gw = Wbase + (size_t)row * K + kt + c0;
#pragma unroll
            for (int q = 0; q < 4; ++q) {
                *(ushort8*)(&As[cur][row * 64 + c0 + q * 8]) = *(const ushort8*)(ga + q * 8);
                *(ushort8*)(&Bs[cur][row * 64 + c0 + q * 8]) = *(const ushort8*)(gw + q * 8);
            }
        }
        __syncthreads();
#endif

#pragma unroll
        for (int ks = 0; ks < 2; ++ks) {
            FragU af[4], bf[2];
#pragma unroll
            for (int mi = 0; mi < 4; ++mi) {
                const int row = wm + mi * 16 + lr;
                const int kOff = ks * 32 + hi * 8;   // A: lo lanes K0..7/16..23
                af[mi].us[0] = *(const ushort8*)(&As[cur][row * 64 + kOff]);
                af[mi].us[1] = *(const ushort8*)(&As[cur][row * 64 + kOff + 16]);
            }
#pragma unroll
            for (int ni = 0; ni < 2; ++ni) {
                const int col = wn + ni * 16 + lr;
                const int kOff = ks * 32 + hi * 8;   // B: lo lanes K0..7/16..23
                bf[ni].us[0] = *(const ushort8*)(&Bs[cur][col * 64 + kOff]);
                bf[ni].us[1] = *(const ushort8*)(&Bs[cur][col * 64 + kOff + 16]);
            }
#pragma unroll
            for (int mi = 0; mi < 4; ++mi)
#pragma unroll
                for (int ni = 0; ni < 2; ++ni)
                    acc[mi][ni] = wmma_bf16(af[mi].bf, bf[ni].bf, acc[mi][ni]);
        }
    }

    // Epilogue: C/D layout -> lane lr is column, rows r + 8*hi.
#pragma unroll
    for (int ni = 0; ni < 2; ++ni) {
        const int n = N0 + wn + ni * 16 + lr;
        const float bv = bias[n];
#pragma unroll
        for (int mi = 0; mi < 4; ++mi) {
            const int mBase = M0 + wm + mi * 16 + hi * 8;
#pragma unroll
            for (int r = 0; r < 8; ++r) {
                float v = acc[mi][ni][r] + bv;
                if (relu) v = fmaxf(v, 0.0f);
                const size_t idx = (size_t)(mBase + r) * N + n;
                if (outF) outF[idx] = v;
                if (outB) outB[idx] = f32_to_bf16(v);
            }
        }
    }
}

// ---------------------------------------------------------------------------
// Flash attention: one block = (b,h) x 64 query rows; 4 waves x 16 q rows.
// KV chunk = 32 keys. scale = 1/sqrt(64) = 0.125.
// ---------------------------------------------------------------------------
__global__ __launch_bounds__(128) void attn_kernel(
        const unsigned short* __restrict__ Qb,
        const unsigned short* __restrict__ Kb,
        const unsigned short* __restrict__ Vb,
        unsigned short* __restrict__ ctx) {
    __shared__ __align__(16) unsigned short Ks[32 * 64];   // [key][d]
    __shared__ __align__(16) unsigned short Vt[64 * 32];   // [d][key] (transposed)
    __shared__ __align__(16) float          scoreBuf[4][16 * 32];
    __shared__ __align__(16) unsigned short pBuf[4][16 * 32];
    __shared__ float rowMax[4][16], rowSum[4][16], rowCorr[4][16];

    const int wave = threadIdx.x >> 5;
    const int lane = threadIdx.x & 31;
    const int lr   = lane & 15;
    const int hi   = lane >> 4;

    const int qt = blockIdx.x;             // 0..31 (64-row q tiles)
    const int bh = blockIdx.y;             // 0..63
    const int b  = bh >> 4;
    const int h  = bh & 15;

    const size_t rowBase = (size_t)b * SS;
    const int    colBase = h * DHH;
    const int    q0      = qt * 64 + wave * 16;

    // Load Q fragments (16 q rows x 64 d) straight from global.
    FragU qf[2];
    {
        const unsigned short* qp = Qb + (rowBase + q0 + lr) * DD + colBase;
#pragma unroll
        for (int c = 0; c < 2; ++c) {
            const int off = c * 32 + hi * 8;
            qf[c].us[0] = *(const ushort8*)(qp + off);
            qf[c].us[1] = *(const ushort8*)(qp + off + 16);
        }
    }

    v8f acc[4];
#pragma unroll
    for (int nt = 0; nt < 4; ++nt)
#pragma unroll
        for (int r = 0; r < 8; ++r) acc[nt][r] = 0.0f;

    if (lane < 16) {
        rowMax[wave][lane] = -1e30f;
        rowSum[wave][lane] = 0.0f;
    }

    for (int k0 = 0; k0 < SS; k0 += 32) {
        __syncthreads();
        // Stage K chunk [32,64] and V chunk transposed [64,32] into LDS.
        {
            const int r    = threadIdx.x >> 2;       // 0..31 key row
            const int cseg = (threadIdx.x & 3) * 16; // d segment
            const unsigned short* kp = Kb + (rowBase + k0 + r) * DD + colBase + cseg;
            *(ushort8*)(Ks + r * 64 + cseg)     = *(const ushort8*)(kp);
            *(ushort8*)(Ks + r * 64 + cseg + 8) = *(const ushort8*)(kp + 8);
            const unsigned short* vp = Vb + (rowBase + k0 + r) * DD + colBase + cseg;
            const ushort8 v0 = *(const ushort8*)(vp);
            const ushort8 v1 = *(const ushort8*)(vp + 8);
#pragma unroll
            for (int j = 0; j < 8; ++j) {
                Vt[(cseg + j) * 32 + r]     = v0[j];
                Vt[(cseg + 8 + j) * 32 + r] = v1[j];
            }
        }
        __syncthreads();

        // Scores: two 16-key tiles, K-dim = 64 (2 WMMAs each).
#pragma unroll
        for (int hf = 0; hf < 2; ++hf) {
            v8f sc;
#pragma unroll
            for (int r = 0; r < 8; ++r) sc[r] = 0.0f;
            const int kcol = hf * 16 + lr;
#pragma unroll
            for (int c = 0; c < 2; ++c) {
                FragU bk;
                const int dOff = c * 32 + hi * 8;
                bk.us[0] = *(const ushort8*)(Ks + kcol * 64 + dOff);
                bk.us[1] = *(const ushort8*)(Ks + kcol * 64 + dOff + 16);
                sc = wmma_bf16(qf[c].bf, bk.bf, sc);
            }
#pragma unroll
            for (int r = 0; r < 8; ++r)
                scoreBuf[wave][(r + hi * 8) * 32 + hf * 16 + lr] = sc[r] * 0.125f;
        }
        __syncthreads();

        // Online softmax over the 32-key chunk (lanes 0..15 own one q row each).
        if (lane < 16) {
            const int qr = lane;
            float mOld = rowMax[wave][qr];
            float cm = -1e30f;
#pragma unroll
            for (int j = 0; j < 32; ++j) cm = fmaxf(cm, scoreBuf[wave][qr * 32 + j]);
            const float mNew = fmaxf(mOld, cm);
            const float corr = __expf(mOld - mNew);
            float s = 0.0f;
#pragma unroll
            for (int j = 0; j < 32; ++j) {
                const float p = __expf(scoreBuf[wave][qr * 32 + j] - mNew);
                s += p;
                pBuf[wave][qr * 32 + j] = f32_to_bf16(p);
            }
            rowSum[wave][qr]  = rowSum[wave][qr] * corr + s;
            rowMax[wave][qr]  = mNew;
            rowCorr[wave][qr] = corr;
        }
        __syncthreads();

        // Rescale accumulators and add P @ V for this chunk.
        {
            FragU pf;
            const int pOff = hi * 8;
            pf.us[0] = *(const ushort8*)(pBuf[wave] + lr * 32 + pOff);
            pf.us[1] = *(const ushort8*)(pBuf[wave] + lr * 32 + pOff + 16);

            float cr[8];
#pragma unroll
            for (int r = 0; r < 8; ++r) cr[r] = rowCorr[wave][r + hi * 8];

#pragma unroll
            for (int nt = 0; nt < 4; ++nt) {
                FragU vf;
                const int dcol = nt * 16 + lr;
                const int kOff = hi * 8;
                vf.us[0] = *(const ushort8*)(Vt + dcol * 32 + kOff);
                vf.us[1] = *(const ushort8*)(Vt + dcol * 32 + kOff + 16);
#pragma unroll
                for (int r = 0; r < 8; ++r) acc[nt][r] *= cr[r];
                acc[nt] = wmma_bf16(pf.bf, vf.bf, acc[nt]);
            }
        }
    }

    __syncthreads();
    float inv[8];
#pragma unroll
    for (int r = 0; r < 8; ++r) inv[r] = 1.0f / rowSum[wave][r + hi * 8];

#pragma unroll
    for (int nt = 0; nt < 4; ++nt) {
        const int dcol = nt * 16 + lr;
#pragma unroll
        for (int r = 0; r < 8; ++r) {
            const int q = q0 + r + hi * 8;
            ctx[(rowBase + q) * DD + colBase + dcol] = f32_to_bf16(acc[nt][r] * inv[r]);
        }
    }
}

// ---------------------------------------------------------------------------
// Residual add + LayerNorm. One block per row (256 threads, 4 elems each).
// ---------------------------------------------------------------------------
__global__ __launch_bounds__(256) void add_ln_kernel(
        const float* __restrict__ x, const float* __restrict__ y,
        const float* __restrict__ gamma, const float* __restrict__ beta,
        float* __restrict__ outF, unsigned short* __restrict__ outB) {
    __shared__ float red[256];
    const int row = blockIdx.x;
    const int t   = threadIdx.x;
    const float* xr = x + (size_t)row * DD;
    const float* yr = y + (size_t)row * DD;

    float v[4];
    float s = 0.0f;
#pragma unroll
    for (int i = 0; i < 4; ++i) {
        v[i] = xr[t * 4 + i] + yr[t * 4 + i];
        s += v[i];
    }
    red[t] = s;
    __syncthreads();
    for (int off = 128; off > 0; off >>= 1) {
        if (t < off) red[t] += red[t + off];
        __syncthreads();
    }
    const float mu = red[0] * (1.0f / DD);
    __syncthreads();

    float sq = 0.0f;
#pragma unroll
    for (int i = 0; i < 4; ++i) {
        const float d = v[i] - mu;
        sq += d * d;
    }
    red[t] = sq;
    __syncthreads();
    for (int off = 128; off > 0; off >>= 1) {
        if (t < off) red[t] += red[t + off];
        __syncthreads();
    }
    const float rs = rsqrtf(red[0] * (1.0f / DD) + EPS);

#pragma unroll
    for (int i = 0; i < 4; ++i) {
        const int c = t * 4 + i;
        const float o = (v[i] - mu) * rs * gamma[c] + beta[c];
        const size_t idx = (size_t)row * DD + c;
        outF[idx] = o;
        if (outB) outB[idx] = f32_to_bf16(o);
    }
}

// ---------------------------------------------------------------------------
// Host launcher
// ---------------------------------------------------------------------------
extern "C" void kernel_launch(void* const* d_in, const int* in_sizes, int n_in,
                              void* d_out, int out_size, void* d_ws, size_t ws_size,
                              hipStream_t stream) {
    const float* src   = (const float*)d_in[0];
    const float* WQ_w  = (const float*)d_in[2];
    const float* WQ_b  = (const float*)d_in[3];
    const float* WK_w  = (const float*)d_in[4];
    const float* WK_b  = (const float*)d_in[5];
    const float* WV_w  = (const float*)d_in[6];
    const float* WV_b  = (const float*)d_in[7];
    const float* WO_w  = (const float*)d_in[8];
    const float* WO_b  = (const float*)d_in[9];
    const float* l1_w  = (const float*)d_in[10];
    const float* l1_b  = (const float*)d_in[11];
    const float* l2_w  = (const float*)d_in[12];
    const float* l2_b  = (const float*)d_in[13];
    const float* ln1_g = (const float*)d_in[14];
    const float* ln1_b = (const float*)d_in[15];
    const float* ln2_g = (const float*)d_in[16];
    const float* ln2_b = (const float*)d_in[17];
    (void)in_sizes; (void)n_in; (void)out_size; (void)ws_size;

    // Bump allocator over workspace (256B aligned chunks).
    char* wp = (char*)d_ws;
    auto alloc = [&](size_t bytes) -> void* {
        void* r = (void*)wp;
        wp += (bytes + 255) & ~(size_t)255;
        return r;
    };
    unsigned short* srcb = (unsigned short*)alloc((size_t)MM * DD * 2);
    unsigned short* wqb  = (unsigned short*)alloc((size_t)DD * DD * 2);
    unsigned short* wkb  = (unsigned short*)alloc((size_t)DD * DD * 2);
    unsigned short* wvb  = (unsigned short*)alloc((size_t)DD * DD * 2);
    unsigned short* wob  = (unsigned short*)alloc((size_t)DD * DD * 2);
    unsigned short* l1b  = (unsigned short*)alloc((size_t)FFN * DD * 2);
    unsigned short* l2b  = (unsigned short*)alloc((size_t)DD * FFN * 2);
    unsigned short* Qb   = (unsigned short*)alloc((size_t)MM * DD * 2);
    unsigned short* Kb   = (unsigned short*)alloc((size_t)MM * DD * 2);
    unsigned short* Vb   = (unsigned short*)alloc((size_t)MM * DD * 2);
    unsigned short* ctxb = (unsigned short*)alloc((size_t)MM * DD * 2);
    float*          attnF= (float*)alloc((size_t)MM * DD * 4);
    float*          x1F  = (float*)alloc((size_t)MM * DD * 4);
    unsigned short* x1b  = (unsigned short*)alloc((size_t)MM * DD * 2);
    unsigned short* hb   = (unsigned short*)alloc((size_t)MM * FFN * 2);
    float*          ffnF = (float*)alloc((size_t)MM * DD * 4);

    auto cast = [&](const float* in, unsigned short* out, int n) {
        cast_bf16_kernel<<<(n + 255) / 256, 256, 0, stream>>>(in, out, n);
    };
    cast(src,  srcb, MM * DD);
    cast(WQ_w, wqb,  DD * DD);
    cast(WK_w, wkb,  DD * DD);
    cast(WV_w, wvb,  DD * DD);
    cast(WO_w, wob,  DD * DD);
    cast(l1_w, l1b,  FFN * DD);
    cast(l2_w, l2b,  DD * FFN);

    // Q/K/V projections: [8192,1024] x [1024,1024]^T
    dim3 gQKV(DD / 128, MM / 128);
    gemm_bf16_kernel<<<gQKV, 256, 0, stream>>>(srcb, wqb, WQ_b, nullptr, Qb, MM, DD, DD, 0);
    gemm_bf16_kernel<<<gQKV, 256, 0, stream>>>(srcb, wkb, WK_b, nullptr, Kb, MM, DD, DD, 0);
    gemm_bf16_kernel<<<gQKV, 256, 0, stream>>>(srcb, wvb, WV_b, nullptr, Vb, MM, DD, DD, 0);

    // Attention: grid (q-tiles of 64, B*H)
    attn_kernel<<<dim3(SS / 64, BB * HH), 128, 0, stream>>>(Qb, Kb, Vb, ctxb);

    // Output projection -> fp32 for residual
    gemm_bf16_kernel<<<gQKV, 256, 0, stream>>>(ctxb, wob, WO_b, attnF, nullptr, MM, DD, DD, 0);

    // x1 = LN(src + attn_out)
    add_ln_kernel<<<MM, 256, 0, stream>>>(src, attnF, ln1_g, ln1_b, x1F, x1b);

    // FFN1: relu(x1 @ l1^T + b1) -> bf16
    gemm_bf16_kernel<<<dim3(FFN / 128, MM / 128), 256, 0, stream>>>(
        x1b, l1b, l1_b, nullptr, hb, MM, FFN, DD, 1);

    // FFN2: h @ l2^T + b2 -> fp32
    gemm_bf16_kernel<<<dim3(DD / 128, MM / 128), 256, 0, stream>>>(
        hb, l2b, l2_b, ffnF, nullptr, MM, DD, FFN, 0);

    // out = LN(x1 + ffn)
    add_ln_kernel<<<MM, 256, 0, stream>>>(x1F, ffnF, ln2_g, ln2_b, (float*)d_out, nullptr);
}
